// BinarizeConv2d_block_9174050144490
// MI455X (gfx1250) — matrix operand order repaired and checked by reading, not compiled, not run on hardware
//
#include <hip/hip_runtime.h>

typedef _Float16 half_t;
typedef __attribute__((ext_vector_type(16))) _Float16 v16h;
typedef __attribute__((ext_vector_type(8)))  float    v8f;

#define CIN   128
#define COUT  256
#define HWDIM 56
#define NPIX  3136      // 56*56
#define KSZ   1152      // 128*9
#define KSTEP 32
#define NKIT  36        // 1152/32
#define NTILE 64
#define LDSTR 40        // halfs per LDS row (32 data + 8 pad), 80B rows (16B aligned)

// ---------------------------------------------------------------------------
// CDNA5 async copy: global -> LDS, 16B per lane, tracked by ASYNCcnt.
// GVS addressing: mem = SGPR64 base + VGPR32 byte offset.  vdst = LDS byte addr.
// ---------------------------------------------------------------------------
__device__ __forceinline__ void async_ld_b128(unsigned lds_addr, unsigned long long gbase,
                                              unsigned goff) {
    asm volatile("global_load_async_to_lds_b128 %0, %1, %2"
                 :: "v"(lds_addr), "v"(goff), "s"(gbase)
                 : "memory");
}
__device__ __forceinline__ void wait_async0() {
    asm volatile("s_wait_asynccnt 0x0" ::: "memory");
}

// ---------------------------------------------------------------------------
// Kernel 1: quantize weights to 2 bits (clamp[-1,1], round(x*2)/2) and pack
// into A-matrix layout aq[co][k] (K-contiguous f16, exact for 2-bit levels).
// ---------------------------------------------------------------------------
__global__ __launch_bounds__(256) void quant_pack_weights(
    const float* __restrict__ w, half_t* __restrict__ aq)
{
    int idx = blockIdx.x * 256 + threadIdx.x;
    if (idx >= COUT * KSZ) return;
    float v = w[idx];                    // OIHW flat == co*1152 + k, same order
    v = fminf(1.f, fmaxf(-1.f, v));
    v = rintf(v * 2.f) * 0.5f;
    aq[idx] = (half_t)v;
}

// ---------------------------------------------------------------------------
// Kernel 2: fold BN running stats into per-channel scale/shift.
// ---------------------------------------------------------------------------
__global__ __launch_bounds__(256) void bn_fold(
    const float* __restrict__ gamma, const float* __restrict__ beta,
    const float* __restrict__ mean,  const float* __restrict__ var,
    float* __restrict__ scb, float* __restrict__ shb)
{
    int i = threadIdx.x;                 // COUT == 256 == blockDim
    float s = gamma[i] * rsqrtf(var[i] + 1e-5f);
    scb[i] = s;
    shb[i] = beta[i] - mean[i] * s;
}

// ---------------------------------------------------------------------------
// Kernel 3: implicit-GEMM conv + BN + hardtanh + 2-bit quant via WMMA f16.
// 256 threads (8 waves), tile = 256 Cout x 64 positions, double-buffered LDS.
// Per K-step per wave: 8x v_wmma_f32_16x16x32_f16; next chunk's A staged by
// global_load_async_to_lds_b128 and B gather loads issued before the WMMAs.
// ---------------------------------------------------------------------------
__global__ __launch_bounds__(256) void conv_bn_quant_wmma(
    const float* __restrict__ x, const half_t* __restrict__ aq,
    const float* __restrict__ scb, const float* __restrict__ shb,
    float* __restrict__ out)
{
    __shared__ __align__(16) half_t Asm[2][COUT * LDSTR];   // 2 x 20480 B
    __shared__ __align__(16) half_t Bsm[2][NTILE * LDSTR];  // 2 x  5120 B

    const int t    = threadIdx.x;
    const int lane = t & 31;
    const int wv   = t >> 5;
    const int wm   = wv >> 1;     // 0..3 -> M fragments (per 128-row half-tile)
    const int wn   = wv & 1;      // 0..1 -> N fragments
    const int lhi  = lane >> 4;
    const int l15  = lane & 15;

    const int pos_base = blockIdx.x * NTILE;
    const int n        = blockIdx.z;
    const float* xn = x + (size_t)n * CIN * NPIX;

    // ---- per-thread im2col slice: fixed pos, 8 consecutive k ----
    const int pos  = t & 63;
    const int kkb  = (t >> 6) * 8;            // 0,8,16,24
    const int p    = pos_base + pos;
    const int h    = p / HWDIM;
    const int w    = p - h * HWDIM;
    const int base = h * HWDIM + w - 57;      // + ci*NPIX + kh*56 + kw
    const int hmask = (h > 0 ? 1 : 0) | 2 | (h < HWDIM - 1 ? 4 : 0);
    const int wmask = (w > 0 ? 1 : 0) | 2 | (w < HWDIM - 1 ? 4 : 0);
    int ci0 = kkb / 9, r0 = kkb - ci0 * 9;    // chunk-start (ci, r), carried

    // ---- async A staging: thread t owns weight row (= Cout) t ----
    const unsigned lds_arow0 = (unsigned)(uintptr_t)(&Asm[0][t * LDSTR]);
    const unsigned lds_arow1 = (unsigned)(uintptr_t)(&Asm[1][t * LDSTR]);
    const unsigned long long abase = (unsigned long long)(uintptr_t)aq;
    unsigned agoff = (unsigned)(t * KSZ) * 2u;

    v8f acc[2][2][2];
    #pragma unroll
    for (int c = 0; c < 2; ++c)
        #pragma unroll
        for (int a = 0; a < 2; ++a)
            #pragma unroll
            for (int b = 0; b < 2; ++b)
                acc[c][a][b] = v8f{};

    // ---- prologue: stage chunk 0 into buffer 0 ----
    async_ld_b128(lds_arow0 +  0, abase, agoff +  0);
    async_ld_b128(lds_arow0 + 16, abase, agoff + 16);
    async_ld_b128(lds_arow0 + 32, abase, agoff + 32);
    async_ld_b128(lds_arow0 + 48, abase, agoff + 48);
    agoff += KSTEP * 2;
    {
        union { half_t hh[8]; uint4 q; } vals;
        int ci = ci0, r = r0;
        #pragma unroll
        for (int j = 0; j < 8; ++j) {
            int kh = r / 3, kw = r - kh * 3;
            int ok = ((hmask >> kh) & 1) & ((wmask >> kw) & 1);
            int off = ci * NPIX + base + kh * HWDIM + kw;
            float v = xn[ok ? off : 0];
            vals.hh[j] = (half_t)(ok ? v : 0.f);
            int nr = r + 1, cw = (nr == 9); ci += cw; r = cw ? 0 : nr;
        }
        *(uint4*)(&Bsm[0][pos * LDSTR + kkb]) = vals.q;
    }
    ci0 += 3; r0 += 5; if (r0 >= 9) { r0 -= 9; ci0 += 1; }

    int buf = 0;
    #pragma unroll 1
    for (int i = 0; i < NKIT; ++i) {
        wait_async0();            // this buffer's async A writes are done
        __syncthreads();          // ...and visible; prior readers of buf^1 done

        const bool more = (i + 1 < NKIT);
        float bv[8];
        int okm = 0;
        if (more) {
            // issue next chunk's A (async -> LDS buf^1) and B gather loads (to VGPRs)
            unsigned ar = buf ? lds_arow0 : lds_arow1;
            async_ld_b128(ar +  0, abase, agoff +  0);
            async_ld_b128(ar + 16, abase, agoff + 16);
            async_ld_b128(ar + 32, abase, agoff + 32);
            async_ld_b128(ar + 48, abase, agoff + 48);
            agoff += KSTEP * 2;
            __builtin_prefetch(aq + (size_t)t * KSZ + (i + 2) * KSTEP, 0, 3);
            int ci = ci0, r = r0;
            #pragma unroll
            for (int j = 0; j < 8; ++j) {
                int kh = r / 3, kw = r - kh * 3;
                int ok = ((hmask >> kh) & 1) & ((wmask >> kw) & 1);
                int off = ci * NPIX + base + kh * HWDIM + kw;
                bv[j] = xn[ok ? off : 0];
                okm |= ok << j;
                int nr = r + 1, cw = (nr == 9); ci += cw; r = cw ? 0 : nr;
            }
            ci0 += 3; r0 += 5; if (r0 >= 9) { r0 -= 9; ci0 += 1; }
        }

        // ---- compute on current buffer: fragments + 8 WMMAs ----
        union V16 { v16h v; uint4 q[2]; };
        const int ak0 = lhi ? 8 : 0;
        const int bk0 = lhi ? 8 : 0;
        const half_t* Ab = Asm[buf];
        const half_t* Bb = Bsm[buf];

        V16 bfrag[2];
        #pragma unroll
        for (int b = 0; b < 2; ++b) {
            const uint4* bp = (const uint4*)(Bb + ((2 * wn + b) * 16 + l15) * LDSTR + bk0);
            bfrag[b].q[0] = bp[0];
            bfrag[b].q[1] = bp[2];
        }
        #pragma unroll
        for (int c = 0; c < 2; ++c) {
            V16 afrag[2];
            #pragma unroll
            for (int a = 0; a < 2; ++a) {
                const uint4* ap = (const uint4*)(Ab + (c * 128 + (2 * wm + a) * 16 + l15) * LDSTR);
                afrag[a].q[0] = ap[ak0 >> 3];
                afrag[a].q[1] = ap[(ak0 + 16) >> 3];
            }
            #pragma unroll
            for (int a = 0; a < 2; ++a)
                #pragma unroll
                for (int b = 0; b < 2; ++b)
                    acc[c][a][b] = __builtin_amdgcn_wmma_f32_16x16x32_f16(
                        false, afrag[a].v, false, bfrag[b].v,
                        (short)0, acc[c][a][b], false, false);
        }

        // ---- finish next chunk's B: convert + one ds_store_b128 into buf^1 ----
        if (more) {
            union { half_t hh[8]; uint4 q; } vals;
            #pragma unroll
            for (int j = 0; j < 8; ++j)
                vals.hh[j] = (half_t)(((okm >> j) & 1) ? bv[j] : 0.f);
            *(uint4*)(&Bsm[buf ^ 1][pos * LDSTR + kkb]) = vals.q;
        }
        buf ^= 1;
    }

    // ---- epilogue: folded BN + hardtanh + 2-bit activation quant ----
    #pragma unroll
    for (int c = 0; c < 2; ++c) {
        #pragma unroll
        for (int a = 0; a < 2; ++a) {
            int co0 = c * 128 + (2 * wm + a) * 16 + lhi * 8;   // multiple of 8
            float4 s0 = *(const float4*)(scb + co0);
            float4 s1 = *(const float4*)(scb + co0 + 4);
            float4 h0 = *(const float4*)(shb + co0);
            float4 h1 = *(const float4*)(shb + co0 + 4);
            float sc[8] = {s0.x, s0.y, s0.z, s0.w, s1.x, s1.y, s1.z, s1.w};
            float sh[8] = {h0.x, h0.y, h0.z, h0.w, h1.x, h1.y, h1.z, h1.w};
            #pragma unroll
            for (int b = 0; b < 2; ++b) {
                int pp = pos_base + (2 * wn + b) * 16 + l15;
                float* op = out + (size_t)n * COUT * NPIX + pp;
                #pragma unroll
                for (int i = 0; i < 8; ++i) {
                    float y = acc[c][a][b][i] * sc[i] + sh[i];
                    y = fminf(1.f, fmaxf(-1.f, y));
                    y = rintf(y * 2.f) * 0.5f;          // AB=2 quantizer (RNE)
                    op[(size_t)(co0 + i) * NPIX] = y;
                }
            }
        }
    }
}

extern "C" void kernel_launch(void* const* d_in, const int* in_sizes, int n_in,
                              void* d_out, int out_size, void* d_ws, size_t ws_size,
                              hipStream_t stream)
{
    const float* x     = (const float*)d_in[0];
    const float* w     = (const float*)d_in[1];
    const float* gamma = (const float*)d_in[2];
    const float* beta  = (const float*)d_in[3];
    const float* mean  = (const float*)d_in[4];
    const float* var   = (const float*)d_in[5];
    float*  out = (float*)d_out;

    half_t* aq  = (half_t*)d_ws;                         // 576 KB
    float*  scb = (float*)((char*)d_ws + (size_t)COUT * KSZ * sizeof(half_t));
    float*  shb = scb + COUT;

    int batch = in_sizes[0] / (CIN * NPIX);              // 64

    quant_pack_weights<<<(COUT * KSZ + 255) / 256, 256, 0, stream>>>(w, aq);
    bn_fold<<<1, 256, 0, stream>>>(gamma, beta, mean, var, scb, shb);

    dim3 grid(NPIX / NTILE, 1, batch);                   // (49, 1, 64)
    conv_bn_quant_wmma<<<grid, 256, 0, stream>>>(x, aq, scb, shb, out);
}
